// MixedPrecisionQATLinearEnhanced_36739150250442
// MI455X (gfx1250) — matrix operand order repaired and checked by reading, not compiled, not run on hardware
//
#include <hip/hip_runtime.h>
#include <stdint.h>

// ---------------------------------------------------------------------------
// MI455X (gfx1250) mixed-precision QAT linear:
//   pass1: weight fake-quant blend  -> bf16 Wq (ws)
//   pass2: activation fake-quant    -> bf16 Xq (ws)
//   pass3: bf16 WMMA GEMM + bias    -> f32 out
//          (double-buffered LDS, BK=64, padded conflict-free LDS rows)
// ---------------------------------------------------------------------------

typedef __attribute__((ext_vector_type(16))) __bf16 bf16x16;
typedef __attribute__((ext_vector_type(8)))  float  f32x8;

#define IN_F   1024
#define OUT_F  4096
#define M_TOK  16384   // 8 * 2048

#define BM 128
#define BN 128
#define BK 64
#define KSTEPS (IN_F / BK)        // 16
#define LDS_STRIDE (BK + 8)       // 72 halfs: 36 words, conflict-free padding

__device__ __constant__ float NF4_LEVELS[16] = {
    -1.0f, -0.6961928009986877f, -0.5250730514526367f, -0.39491748809814453f,
    -0.28444138169288635f, -0.18477343022823334f, -0.09105003625154495f, 0.0f,
    0.07958029955625534f, 0.16093020141124725f, 0.24611230194568634f,
    0.33791524171829224f, 0.44070982933044434f, 0.5626170039176941f,
    0.7229568362236023f, 1.0f};

__device__ __constant__ float NF4_BOUNDS[15] = {
    -0.8480964004993439f, -0.6106329262256622f, -0.4599952697753906f,
    -0.33967943489551544f, -0.23460740596055984f, -0.13791173323988914f,
    -0.045525018125772476f, 0.03979014977812767f, 0.1202552504837513f,
    0.2035212516784668f, 0.2920137718319893f, 0.3893125355243683f,
    0.5016634166240692f, 0.6427869200706482f, 0.8614784181118011f};

__device__ __forceinline__ unsigned short f32_to_bf16_rne(float f) {
    uint32_t u = __builtin_bit_cast(uint32_t, f);
    uint32_t r = u + 0x7FFFu + ((u >> 16) & 1u);
    return (unsigned short)(r >> 16);
}

__device__ __forceinline__ void softmax3(const float* __restrict__ l,
                                         float& p0, float& p1, float& p2) {
    float a = l[0], b = l[1], c = l[2];
    float mx = fmaxf(a, fmaxf(b, c));
    float e0 = __expf(a - mx), e1 = __expf(b - mx), e2 = __expf(c - mx);
    float inv = 1.0f / (e0 + e1 + e2);
    p0 = e0 * inv; p1 = e1 * inv; p2 = e2 * inv;
}

// ---------------------------------------------------------------------------
// Pass 1: weight quant blend.  One wave32 per NF4 block of 64 elements.
// ---------------------------------------------------------------------------
__global__ __launch_bounds__(256) void qat_prep_w(
    const float* __restrict__ W, const float* __restrict__ lsq_s,
    const float* __restrict__ logits_w, unsigned short* __restrict__ Wq) {
    const int lane  = threadIdx.x & 31;
    const int wave  = threadIdx.x >> 5;
    const int blk   = blockIdx.x * 8 + wave;       // 64-elem block id
    const int base  = blk * 64 + lane * 2;

    float2 w2 = *(const float2*)(W + base);

    float p0, p1, p2;
    softmax3(logits_w, p0, p1, p2);

    float am = fmaxf(fabsf(w2.x), fabsf(w2.y));
#pragma unroll
    for (int off = 16; off >= 1; off >>= 1)
        am = fmaxf(am, __shfl_xor(am, off, 32));
    am = fmaxf(am, 1e-8f);
    const float inv_am = 1.0f / am;

    const int row = blk >> 4;                      // 16 blocks per 1024-row
    const float s = lsq_s[row];
    const float inv_s = 1.0f / s;

    float wv[2] = {w2.x, w2.y};
    float outv[2];
#pragma unroll
    for (int i = 0; i < 2; ++i) {
        float t = wv[i] * inv_am;
        int idx = 0;
#pragma unroll
        for (int j = 0; j < 15; ++j) idx += (t > NF4_BOUNDS[j]) ? 1 : 0;
        float nf4 = NF4_LEVELS[idx] * am;
        float tl  = fminf(fmaxf(wv[i] * inv_s, -128.0f), 127.0f);
        float lsq = rintf(tl) * s;
        outv[i] = p0 * nf4 + p1 * lsq + p2 * wv[i];
    }
    unsigned int packed = (unsigned int)f32_to_bf16_rne(outv[0]) |
                          ((unsigned int)f32_to_bf16_rne(outv[1]) << 16);
    *(unsigned int*)(Wq + base) = packed;
}

// ---------------------------------------------------------------------------
// Pass 2: activation quant blend.  One 256-thread WG per token row.
// ---------------------------------------------------------------------------
__global__ __launch_bounds__(256) void qat_prep_x(
    const float* __restrict__ X, const float* __restrict__ logits_a,
    const float* __restrict__ alpha_p, const float* __restrict__ beta_p,
    unsigned short* __restrict__ Xq) {
    __shared__ float red_mn[8];
    __shared__ float red_mx[8];

    const int row  = blockIdx.x;
    const int tid  = threadIdx.x;
    const int lane = tid & 31;
    const int wave = tid >> 5;

    const float* xr = X + (size_t)row * IN_F;
    float4 v = *(const float4*)(xr + tid * 4);

    float mn = fminf(fminf(v.x, v.y), fminf(v.z, v.w));
    float mx = fmaxf(fmaxf(v.x, v.y), fmaxf(v.z, v.w));
#pragma unroll
    for (int off = 16; off >= 1; off >>= 1) {
        mn = fminf(mn, __shfl_xor(mn, off, 32));
        mx = fmaxf(mx, __shfl_xor(mx, off, 32));
    }
    if (lane == 0) { red_mn[wave] = mn; red_mx[wave] = mx; }
    __syncthreads();
    mn = red_mn[0]; mx = red_mx[0];
#pragma unroll
    for (int i = 1; i < 8; ++i) {
        mn = fminf(mn, red_mn[i]);
        mx = fmaxf(mx, red_mx[i]);
    }

    float p0, p1, p2;
    softmax3(logits_a, p0, p1, p2);

    const float alpha = *alpha_p;
    const float beta  = *beta_p;
    const float sc_p  = (alpha - beta) * (1.0f / 255.0f);
    const float isc_p = 1.0f / sc_p;
    const float sc_d  = fmaxf((mx - mn) * (1.0f / 15.0f), 1e-8f);
    const float isc_d = 1.0f / sc_d;

    float in[4]  = {v.x, v.y, v.z, v.w};
    unsigned short ov[4];
#pragma unroll
    for (int i = 0; i < 4; ++i) {
        float x = in[i];
        float xc = fminf(fmaxf(x, beta), alpha);
        float qp = rintf((xc - beta) * isc_p) * sc_p + beta;   // PACT8
        float qd = rintf((x - mn) * isc_d) * sc_d + mn;        // dyn token 4
        ov[i] = f32_to_bf16_rne(p0 * qp + p1 * qd + p2 * x);
    }
    uint2 packed;
    packed.x = (unsigned int)ov[0] | ((unsigned int)ov[1] << 16);
    packed.y = (unsigned int)ov[2] | ((unsigned int)ov[3] << 16);
    *(uint2*)(Xq + (size_t)row * IN_F + tid * 4) = packed;
}

// ---------------------------------------------------------------------------
// Pass 3: bf16 WMMA GEMM, out[M,N] = Xq[M,K] * Wq[N,K]^T + bias.
// WG = 256 threads = 8 waves arranged 4 (M) x 2 (N) over a 128x128 tile.
// Wave owns 32x64: per 64-wide K-slab it runs 2 x (2 A-frags x 4 B-frags)
// = 16 WMMAs between barriers.  Double-buffered LDS, register staging,
// one barrier per K-slab.  LDS rows padded to 72 halfs (36 words) so
// b128 fragment loads are bank-conflict-free.
// Fragment addressing per CDNA5 16-bit WMMA VGPR layouts:
//   A(16x32): lanes 0-15 K={0..7,16..23}, lanes 16-31 K={8..15,24..31}
//   B(32x16): lane%16 = N column, K split same as A
//   C(16x16): vgpr r -> M = r + 8*(lane/16), N = lane%16
// ---------------------------------------------------------------------------
__global__ __launch_bounds__(256) void qat_wmma_gemm(
    const unsigned short* __restrict__ Xq, const unsigned short* __restrict__ Wq,
    const float* __restrict__ bias, float* __restrict__ out) {
    __shared__ __attribute__((aligned(16))) unsigned short Xs[2][BM * LDS_STRIDE]; // 2 x 18 KB
    __shared__ __attribute__((aligned(16))) unsigned short Ws[2][BN * LDS_STRIDE]; // 2 x 18 KB

    const int tid    = threadIdx.x;
    const int lane   = tid & 31;
    const int wave   = tid >> 5;
    const int l16    = lane & 15;
    const int half   = lane >> 4;
    const int wave_m = wave & 3;    // 4 M-strips of 32 rows
    const int wave_n = wave >> 2;   // 2 N-strips of 64 cols

    const int gM = blockIdx.y * BM;
    const int gN = blockIdx.x * BN;

    const unsigned short* Xbase = Xq + (size_t)gM * IN_F;
    const unsigned short* Wbase = Wq + (size_t)gN * IN_F;

    // staging geometry: tile = 128 rows x 64 halfs = 1024 segs of 16B,
    // thread owns 4 segs per array
    int srow[4], scol[4];
#pragma unroll
    for (int i = 0; i < 4; ++i) {
        int seg = tid + i * 256;
        srow[i] = seg >> 3;
        scol[i] = (seg & 7) * 8;
    }

    f32x8 acc[2][4] = {};
    union Frag { uint4 q[2]; bf16x16 v; };

    // ---- preload slab 0 into LDS buffer 0 ----
    {
        uint4 sx[4], sw[4];
#pragma unroll
        for (int i = 0; i < 4; ++i) {
            sx[i] = *(const uint4*)(Xbase + (size_t)srow[i] * IN_F + scol[i]);
            sw[i] = *(const uint4*)(Wbase + (size_t)srow[i] * IN_F + scol[i]);
        }
#pragma unroll
        for (int i = 0; i < 4; ++i) {
            *(uint4*)(&Xs[0][srow[i] * LDS_STRIDE + scol[i]]) = sx[i];
            *(uint4*)(&Ws[0][srow[i] * LDS_STRIDE + scol[i]]) = sw[i];
        }
    }
    __syncthreads();

    for (int kt = 0; kt < KSTEPS; ++kt) {
        const int  cur      = kt & 1;
        const int  nxt      = cur ^ 1;
        const bool has_next = (kt + 1) < KSTEPS;
        const int  koff     = (kt + 1) * BK;

        // ---- stage next slab into registers (latency hidden by WMMAs) ----
        uint4 sx[4], sw[4];
        if (has_next) {
#pragma unroll
            for (int i = 0; i < 4; ++i) {
                sx[i] = *(const uint4*)(Xbase + (size_t)srow[i] * IN_F + koff + scol[i]);
                sw[i] = *(const uint4*)(Wbase + (size_t)srow[i] * IN_F + koff + scol[i]);
            }
        }
        if (kt + 2 < KSTEPS) {   // L2 prefetch two slabs ahead
            __builtin_prefetch(Xbase + (size_t)srow[0] * IN_F + koff + BK + scol[0], 0, 1);
            __builtin_prefetch(Wbase + (size_t)srow[0] * IN_F + koff + BK + scol[0], 0, 1);
        }

        // ---- compute on current buffer: 2 x (2x4) WMMAs ----
        const unsigned short* xs = &Xs[cur][0];
        const unsigned short* ws = &Ws[cur][0];
        const int am0 = (wave_m * 32 + l16) * LDS_STRIDE;
        const int am1 = am0 + 16 * LDS_STRIDE;

#pragma unroll
        for (int kk = 0; kk < 2; ++kk) {
            const int ko = kk * 32;
            Frag a0, a1;
            a0.q[0] = *(const uint4*)(xs + am0 + ko + half * 8);
            a0.q[1] = *(const uint4*)(xs + am0 + ko + half * 8 + 16);
            a1.q[0] = *(const uint4*)(xs + am1 + ko + half * 8);
            a1.q[1] = *(const uint4*)(xs + am1 + ko + half * 8 + 16);

#pragma unroll
            for (int nt = 0; nt < 4; ++nt) {
                Frag b;
                const int bn = (wave_n * 64 + nt * 16 + l16) * LDS_STRIDE + ko;
                b.q[0] = *(const uint4*)(ws + bn + half * 8);
                b.q[1] = *(const uint4*)(ws + bn + half * 8 + 16);
                acc[0][nt] = __builtin_amdgcn_wmma_f32_16x16x32_bf16(
                    false, a0.v, false, b.v, (short)0, acc[0][nt], false, false);
                acc[1][nt] = __builtin_amdgcn_wmma_f32_16x16x32_bf16(
                    false, a1.v, false, b.v, (short)0, acc[1][nt], false, false);
            }
        }

        // ---- commit staged slab to the other buffer, single barrier ----
        if (has_next) {
#pragma unroll
            for (int i = 0; i < 4; ++i) {
                *(uint4*)(&Xs[nxt][srow[i] * LDS_STRIDE + scol[i]]) = sx[i];
                *(uint4*)(&Ws[nxt][srow[i] * LDS_STRIDE + scol[i]]) = sw[i];
            }
            __syncthreads();
        }
    }

    // ---- epilogue: bias add + store per C-matrix layout ----
#pragma unroll
    for (int mi = 0; mi < 2; ++mi) {
#pragma unroll
        for (int nt = 0; nt < 4; ++nt) {
            const int n  = gN + wave_n * 64 + nt * 16 + l16;
            const float bv = bias[n];
#pragma unroll
            for (int r = 0; r < 8; ++r) {
                const int m = gM + wave_m * 32 + mi * 16 + r + half * 8;
                out[(size_t)m * OUT_F + n] = acc[mi][nt][r] + bv;
            }
        }
    }
}

// ---------------------------------------------------------------------------
extern "C" void kernel_launch(void* const* d_in, const int* in_sizes, int n_in,
                              void* d_out, int out_size, void* d_ws, size_t ws_size,
                              hipStream_t stream) {
    const float* x          = (const float*)d_in[0];   // (8,2048,1024)
    const float* weight     = (const float*)d_in[1];   // (4096,1024)
    const float* bias       = (const float*)d_in[2];   // (4096,)
    const float* logits_w   = (const float*)d_in[3];   // (3,)
    const float* logits_a   = (const float*)d_in[4];   // (3,)
    const float* lsq_s      = (const float*)d_in[5];   // (4096,1)
    const float* pact_alpha = (const float*)d_in[6];   // scalar
    const float* pact_beta  = (const float*)d_in[7];   // scalar

    unsigned short* Wq = (unsigned short*)d_ws;                 // 8 MB
    unsigned short* Xq = Wq + (size_t)OUT_F * IN_F;             // 32 MB
    float* out = (float*)d_out;

    qat_prep_w<<<dim3((OUT_F * IN_F / 64) / 8), dim3(256), 0, stream>>>(
        weight, lsq_s, logits_w, Wq);
    qat_prep_x<<<dim3(M_TOK), dim3(256), 0, stream>>>(
        x, logits_a, pact_alpha, pact_beta, Xq);
    qat_wmma_gemm<<<dim3(OUT_F / BN, M_TOK / BM), dim3(256), 0, stream>>>(
        Xq, Wq, bias, out);
}
